// PoPETransformerBlock_71442486001861
// MI455X (gfx1250) — matrix operand order repaired and checked by reading, not compiled, not run on hardware
//
#include <hip/hip_runtime.h>
#include <hip/hip_bf16.h>

typedef __attribute__((ext_vector_type(16))) __bf16 v16bf;
typedef __attribute__((ext_vector_type(8)))  __bf16 v8bf;
typedef __attribute__((ext_vector_type(8)))  float  v8f;

#if __has_builtin(__builtin_amdgcn_tensor_load_to_lds)
#define HAVE_TDM 1
typedef unsigned int u32x4 __attribute__((ext_vector_type(4)));
typedef int          i32x8 __attribute__((ext_vector_type(8)));
typedef int          i32x4 __attribute__((ext_vector_type(4)));

// Issue one 2-D TDM tile load: global (row-major, elem=2B) -> padded LDS.
// g1w* carry data_size/pad config + dims/strides per the D# layout.
static __device__ __forceinline__ void tdm_load_2d(
    unsigned lds_addr, unsigned long long ga,
    int g1w0, int g1w1, int g1w2, int g1w3, int g1w4, int g1w5) {
    u32x4 g0;
    g0[0] = 1u;                                            // count=1, user mode
    g0[1] = lds_addr;                                      // LDS byte address
    g0[2] = (unsigned)ga;                                  // global addr lo
    g0[3] = (unsigned)((ga >> 32) & 0x01FFFFFFu) | 0x80000000u; // hi | type=2
    i32x8 g1 = {g1w0, g1w1, g1w2, g1w3, g1w4, g1w5, 0, 0};
    i32x4 z4 = {0, 0, 0, 0};
#if defined(__clang_major__) && (__clang_major__ >= 23)
    i32x8 z8 = {0, 0, 0, 0, 0, 0, 0, 0};
    __builtin_amdgcn_tensor_load_to_lds(g0, g1, z4, z4, z8, 0);
#else
    __builtin_amdgcn_tensor_load_to_lds(g0, g1, z4, z4, 0);
#endif
}
#else
#define HAVE_TDM 0
#endif

static __device__ __forceinline__ float softplusf(float x) {
    return x > 20.f ? x : log1pf(__expf(x));
}
static __device__ __forceinline__ float gelu_exactf(float x) {
    return 0.5f * x * (1.f + erff(x * 0.70710678118654752f));
}

// ---------------------------------------------------------------------------
// Weight transpose + convert: W[k][n] f32 -> Wt[n][k] bf16 (tiled via LDS).
// grid = (N/32, K/32), block = 256 (32 x 8).
// ---------------------------------------------------------------------------
__global__ __launch_bounds__(256) void cvt_transpose_bf16_kernel(
    const float* __restrict__ src, __bf16* __restrict__ dst, int Kd, int Nd) {
    __shared__ __bf16 tile[32][33];
    const int n0 = blockIdx.x * 32, k0 = blockIdx.y * 32;
    const int tx = threadIdx.x & 31, ty = threadIdx.x >> 5;   // ty = 0..7
#pragma unroll
    for (int i = 0; i < 32; i += 8)
        tile[ty + i][tx] = (__bf16)src[(size_t)(k0 + ty + i) * Nd + n0 + tx];
    __syncthreads();
#pragma unroll
    for (int i = 0; i < 32; i += 8)
        dst[(size_t)(n0 + ty + i) * Kd + k0 + tx] = tile[tx][ty + i];
}

// ---------------------------------------------------------------------------
// LayerNorm (one 256-thread block per row of 1024), writes bf16
// ---------------------------------------------------------------------------
__global__ __launch_bounds__(256) void ln_bf16_kernel(const float* __restrict__ x,
                                                      const float* __restrict__ g,
                                                      const float* __restrict__ b,
                                                      __bf16* __restrict__ out) {
    const int row = blockIdx.x;
    const int tid = threadIdx.x;
    const float* xr = x + (size_t)row * 1024;
    float4 v = ((const float4*)xr)[tid];           // 256 * 4 = 1024
    float s  = v.x + v.y + v.z + v.w;
    float ss = v.x * v.x + v.y * v.y + v.z * v.z + v.w * v.w;
#pragma unroll
    for (int off = 1; off < 32; off <<= 1) {
        s  += __shfl_xor(s,  off, 32);
        ss += __shfl_xor(ss, off, 32);
    }
    __shared__ float rs[8], rss[8];
    int w = tid >> 5, lane = tid & 31;
    if (lane == 0) { rs[w] = s; rss[w] = ss; }
    __syncthreads();
    if (tid == 0) {
        float a = 0.f, c = 0.f;
#pragma unroll
        for (int i = 0; i < 8; ++i) { a += rs[i]; c += rss[i]; }
        rs[0] = a; rss[0] = c;
    }
    __syncthreads();
    float mean = rs[0] * (1.f / 1024.f);
    float var  = rss[0] * (1.f / 1024.f) - mean * mean;
    float inv  = rsqrtf(var + 1e-5f);
    float vals[4] = {v.x, v.y, v.z, v.w};
#pragma unroll
    for (int i = 0; i < 4; ++i) {
        int col = tid * 4 + i;
        float y = (vals[i] - mean) * inv * g[col] + b[col];
        out[(size_t)row * 1024 + col] = (__bf16)y;
    }
}

// ---------------------------------------------------------------------------
// Tiled WMMA GEMM: C[M,N] = A[M,K] * Bt[N,K]^T, bf16 in, f32 accum.
// Block = 256 threads (8 waves), tile 128x128, BK=32.
// A tile: pipelined TDM (double-buffered LDS; wave 0 issues tensor_load_to_lds
// for tile kt+32 before computing tile kt; s_wait_tensorcnt 1 overlaps DMA).
// B tile: pre-transposed weights -> contiguous loads + vector LDS stores.
// epi: 0=none, 2=bias+residual, 3=bias+GELU(exact)
// ---------------------------------------------------------------------------
__global__ __launch_bounds__(256) void gemm_bf16_kernel(
    const __bf16* __restrict__ A, const __bf16* __restrict__ Bt_g,
    float* __restrict__ Cf, __bf16* __restrict__ Cbf,
    const float* __restrict__ bias, const float* __restrict__ resid,
    int M, int N, int K, int epi) {
    __shared__ __align__(16) __bf16 As[2][128][40];  // [m][k], row stride 80B
    __shared__ __align__(16) __bf16 Bt[128][40];     // [n][k], row stride 80B

    const int tid  = threadIdx.x;
    const int w    = tid >> 5, lane = tid & 31;
    const int r    = lane & 15, half = lane >> 4;
    const int wm   = w >> 1, wn = w & 1;
    const int gm   = blockIdx.y * 128, gn = blockIdx.x * 128;

    v8f acc[2][4];
#pragma unroll
    for (int i = 0; i < 2; ++i)
#pragma unroll
        for (int j = 0; j < 4; ++j)
#pragma unroll
            for (int e = 0; e < 8; ++e) acc[i][j][e] = 0.f;

    const int lb_row = tid >> 1, lb_half = tid & 1;      // B tile: 128 rows x 2 halves
    const __bf16* Bptr = Bt_g + (size_t)(gn + lb_row) * K + lb_half * 16;

#if HAVE_TDM
    const unsigned lds_as0 = (unsigned)(unsigned long long)(const void*)&As[0][0][0];
    const unsigned lds_as1 = (unsigned)(unsigned long long)(const void*)&As[1][0][0];
    const unsigned long long abase =
        (unsigned long long)(const void*)(A + (size_t)gm * K);
    // data_size=1(2B) | pad_en | pad_interval=3(16dw) | pad_amount=3(4dw)
    const int g1w0 = (int)0x06D10000u;
    const int g1w1 = (int)((unsigned)(K & 0xFFFF) << 16);              // dim0 lo
    const int g1w2 = (int)(((unsigned)(K >> 16) & 0xFFFFu) |
                           ((unsigned)(M & 0xFFFF) << 16));            // dim0 hi | dim1 lo
    const int g1w3 = (int)(((unsigned)(M >> 16) & 0xFFFFu) |
                           (32u << 16));                               // dim1 hi | tile_dim0
    const int g1w4 = 128;                                              // tile_dim1
    const int g1w5 = K;                                                // dim0_stride
    if (w == 0)
        tdm_load_2d(lds_as0, abase, g1w0, g1w1, g1w2, g1w3, g1w4, g1w5);
#else
    const int la_row = tid >> 1, la_half = tid & 1;
    const __bf16* Aptr = A + (size_t)(gm + la_row) * K + la_half * 16;
#endif

    for (int kt = 0; kt < K; kt += 32) {
        const int buf = (kt >> 5) & 1;
#if HAVE_TDM
        if (w == 0) {
            if (kt + 32 < K) {
                tdm_load_2d(buf ? lds_as0 : lds_as1,
                            abase + (unsigned long long)(kt + 32) * 2ull,
                            g1w0, g1w1, g1w2, g1w3, g1w4, g1w5);
                __builtin_amdgcn_s_wait_tensorcnt(1);
            } else {
                __builtin_amdgcn_s_wait_tensorcnt(0);
            }
        }
#else
        v8bf a0 = *(const v8bf*)(Aptr);
        v8bf a1 = *(const v8bf*)(Aptr + 8);
        *(v8bf*)&As[buf][la_row][la_half * 16]     = a0;
        *(v8bf*)&As[buf][la_row][la_half * 16 + 8] = a1;
        Aptr += 32;
#endif
        v8bf b0 = *(const v8bf*)(Bptr);
        v8bf b1 = *(const v8bf*)(Bptr + 8);
        *(v8bf*)&Bt[lb_row][lb_half * 16]     = b0;
        *(v8bf*)&Bt[lb_row][lb_half * 16 + 8] = b1;
        Bptr += 32;
        if (kt + 32 < K) __builtin_prefetch(Bptr, 0, 3);
        __syncthreads();

        // A frags: lane r = row M; elems e<8 -> K=e+8h, e>=8 -> K=e+8+8h
        v16bf af[2];
#pragma unroll
        for (int i = 0; i < 2; ++i) {
            const __bf16* p = &As[buf][wm * 32 + i * 16 + r][8 * half];
            v8bf lo = *(const v8bf*)p;
            v8bf hi = *(const v8bf*)(p + 16);
#pragma unroll
            for (int e = 0; e < 8; ++e) { af[i][e] = lo[e]; af[i][8 + e] = hi[e]; }
        }
        // B frags: lane r = col N; elems e<8 -> K=e+8h, e>=8 -> K=e+8+8h
#pragma unroll
        for (int j = 0; j < 4; ++j) {
            v16bf bfv;
            const __bf16* p = &Bt[wn * 64 + j * 16 + r][8 * half];
            v8bf lo = *(const v8bf*)p;
            v8bf hi = *(const v8bf*)(p + 16);
#pragma unroll
            for (int e = 0; e < 8; ++e) { bfv[e] = lo[e]; bfv[8 + e] = hi[e]; }
#pragma unroll
            for (int i = 0; i < 2; ++i)
                acc[i][j] = __builtin_amdgcn_wmma_f32_16x16x32_bf16(
                    false, af[i], false, bfv, (short)0, acc[i][j], false, false);
        }
        __syncthreads();
    }

    // epilogue: C layout lane = (N = lane&15, M = vr + 8*(lane>>4))
#pragma unroll
    for (int i = 0; i < 2; ++i)
#pragma unroll
        for (int j = 0; j < 4; ++j) {
            int n = gn + wn * 64 + j * 16 + r;
            float bv = bias ? bias[n] : 0.f;
#pragma unroll
            for (int vr = 0; vr < 8; ++vr) {
                int mrow = gm + wm * 32 + i * 16 + vr + 8 * half;
                float v = acc[i][j][vr] + bv;
                if (epi == 2) v += resid[(size_t)mrow * N + n];
                if (epi == 3) v = gelu_exactf(v);
                if (Cf)  Cf[(size_t)mrow * N + n]  = v;
                if (Cbf) Cbf[(size_t)mrow * N + n] = (__bf16)v;
            }
        }
}

// ---------------------------------------------------------------------------
// PoPE: qkv f32 [B*N, 3*1024] -> q2/k2 bf16 [B,H,N,128],
//       V transposed: vt bf16 [B,H,DH,N]  (keys contiguous -> vector tiles)
// one thread per (b,h,n,d);  idx bits: d:6, n:10, h:4, b:2
// ---------------------------------------------------------------------------
__global__ __launch_bounds__(256) void pope_kernel(const float* __restrict__ qkv,
                                                   const float* __restrict__ phase,
                                                   __bf16* __restrict__ q2,
                                                   __bf16* __restrict__ k2,
                                                   __bf16* __restrict__ vt) {
    int idx = blockIdx.x * 256 + threadIdx.x;
    int d = idx & 63;
    int n = (idx >> 6) & 1023;
    int h = (idx >> 16) & 15;
    int b = idx >> 20;
    size_t qoff = ((size_t)(b * 1024 + n)) * 3072 + h * 64 + d;
    float q = qkv[qoff];
    float k = qkv[qoff + 1024];
    float v = qkv[qoff + 2048];
    float freq = __expf(-(float)d * (9.2103403719761836f / 64.f)); // 10000^(-d/64)
    float th = (float)n * freq;
    float ph = phase[h * 64 + d];
    float mq = softplusf(q), mk = softplusf(k);
    float sq, cq, sk, ck;
    __sincosf(th, &sq, &cq);
    __sincosf(th + ph, &sk, &ck);
    size_t ob = ((size_t)((b * 16 + h) * 1024 + n)) * 128 + d;
    q2[ob]      = (__bf16)(mq * cq);
    q2[ob + 64] = (__bf16)(mq * sq);
    k2[ob]      = (__bf16)(mk * ck);
    k2[ob + 64] = (__bf16)(mk * sk);
    vt[((size_t)((b * 16 + h) * 64 + d)) * 1024 + n] = (__bf16)v;
}

// ---------------------------------------------------------------------------
// Flash attention with WMMA. grid = (N/128, B*H). 8 waves; wave w owns q rows
// [128*blockIdx.x + 16*w, +16). K2 tiles stream through double-buffered LDS
// via pipelined TDM; V tiles come pre-transposed and load with b128 ops.
// o written as [B, N, H, DH] bf16 (matches INNER = h*64+dh ordering).
// ---------------------------------------------------------------------------
__global__ __launch_bounds__(256) void attn_kernel(const __bf16* __restrict__ q2,
                                                   const __bf16* __restrict__ k2,
                                                   const __bf16* __restrict__ vt,
                                                   __bf16* __restrict__ o) {
    const int bh  = blockIdx.y;             // b*16 + h
    const int qt  = blockIdx.x;             // q tile of 128
    const int tid = threadIdx.x;
    const int w   = tid >> 5, lane = tid & 31;
    const int r   = lane & 15, half = lane >> 4;

    __shared__ __align__(16) __bf16 Ks[2][32][136];  // [key][feat 0..127]
    __shared__ __align__(16) __bf16 Vt[64][40];      // [dh][key 0..31]
    __shared__ __align__(16) __bf16 Ps[8][16][40];   // per-wave P staging

    // Q fragments (16 rows x 128 feats = 4 A-frags), straight from global.
    const __bf16* qbase = q2 + ((size_t)bh * 1024 + qt * 128 + w * 16) * 128;
    v16bf qf[4];
#pragma unroll
    for (int c = 0; c < 4; ++c) {
        const __bf16* p = qbase + (size_t)r * 128 + c * 32 + 8 * half;
        v8bf lo = *(const v8bf*)p;
        v8bf hi = *(const v8bf*)(p + 16);
#pragma unroll
        for (int e = 0; e < 8; ++e) { qf[c][e] = lo[e]; qf[c][8 + e] = hi[e]; }
    }

    v8f of[4];
#pragma unroll
    for (int j = 0; j < 4; ++j)
#pragma unroll
        for (int e = 0; e < 8; ++e) of[j][e] = 0.f;
    float m8[8], l8[8];
#pragma unroll
    for (int i = 0; i < 8; ++i) { m8[i] = -1e30f; l8[i] = 0.f; }

#if HAVE_TDM
    const unsigned lds_ks0 = (unsigned)(unsigned long long)(const void*)&Ks[0][0][0];
    const unsigned lds_ks1 = (unsigned)(unsigned long long)(const void*)&Ks[1][0][0];
    const unsigned long long kbase =
        (unsigned long long)(const void*)(k2 + (size_t)bh * 1024 * 128);
    // data_size=1 | pad_en | pad_interval=5(64dw) | pad_amount=3(4dw)
    const int kg1w0 = (int)0x07510000u;
    const int kg1w1 = (int)(128u << 16);     // tensor_dim0 = 128
    const int kg1w2 = (int)(1024u << 16);    // tensor_dim1 = 1024
    const int kg1w3 = (int)(128u << 16);     // tile_dim0 = 128
    const int kg1w4 = 32;                    // tile_dim1 = 32
    const int kg1w5 = 128;                   // dim0_stride = 128
    if (w == 0)
        tdm_load_2d(lds_ks0, kbase, kg1w0, kg1w1, kg1w2, kg1w3, kg1w4, kg1w5);
#else
    const int ldk = tid >> 3, ldc = tid & 7;
#endif
    const int vdh = tid >> 2, vch = tid & 3;     // V tile: 64 dh x 4 chunks of 8

    for (int kt = 0; kt < 1024; kt += 32) {
        const int buf = (kt >> 5) & 1;
#if HAVE_TDM
        if (w == 0) {
            if (kt + 32 < 1024) {
                tdm_load_2d(buf ? lds_ks0 : lds_ks1,
                            kbase + (unsigned long long)(kt + 32) * 256ull,
                            kg1w0, kg1w1, kg1w2, kg1w3, kg1w4, kg1w5);
                __builtin_amdgcn_s_wait_tensorcnt(1);
            } else {
                __builtin_amdgcn_s_wait_tensorcnt(0);
            }
        }
#else
        {
            const __bf16* src = k2 + ((size_t)bh * 1024 + kt + ldk) * 128 + ldc * 16;
            v8bf a = *(const v8bf*)src;
            v8bf b = *(const v8bf*)(src + 8);
            *(v8bf*)&Ks[buf][ldk][ldc * 16]     = a;
            *(v8bf*)&Ks[buf][ldk][ldc * 16 + 8] = b;
        }
#endif
        // V tile: [dh][key] -> vector copy (pre-transposed in global)
        {
            const __bf16* vsrc = vt + ((size_t)bh * 64 + vdh) * 1024 + kt + vch * 8;
            *(v8bf*)&Vt[vdh][vch * 8] = *(const v8bf*)vsrc;
        }
        __syncthreads();

        // --- scores: two 16x16 frags (keys 0-15, 16-31), K=128 in 4 chunks ---
        v8f s0, s1;
#pragma unroll
        for (int e = 0; e < 8; ++e) { s0[e] = 0.f; s1[e] = 0.f; }
#pragma unroll
        for (int c = 0; c < 4; ++c) {
            v16bf b0, b1;
            {
                const __bf16* p = &Ks[buf][r][c * 32 + 8 * half];
                v8bf lo = *(const v8bf*)p;
                v8bf hi = *(const v8bf*)(p + 16);
#pragma unroll
                for (int e = 0; e < 8; ++e) { b0[e] = lo[e]; b0[8 + e] = hi[e]; }
            }
            {
                const __bf16* p = &Ks[buf][16 + r][c * 32 + 8 * half];
                v8bf lo = *(const v8bf*)p;
                v8bf hi = *(const v8bf*)(p + 16);
#pragma unroll
                for (int e = 0; e < 8; ++e) { b1[e] = lo[e]; b1[8 + e] = hi[e]; }
            }
            s0 = __builtin_amdgcn_wmma_f32_16x16x32_bf16(false, qf[c], false, b0,
                                                         (short)0, s0, false, false);
            s1 = __builtin_amdgcn_wmma_f32_16x16x32_bf16(false, qf[c], false, b1,
                                                         (short)0, s1, false, false);
        }

        // --- online softmax (rows distributed: lane holds rows vr+8*half) ---
        float cf[8];
#pragma unroll
        for (int vr = 0; vr < 8; ++vr) {
            float a0 = s0[vr] * 0.125f;   // scale = DH^-0.5
            float a1 = s1[vr] * 0.125f;
            float mx = fmaxf(a0, a1);
#pragma unroll
            for (int off = 1; off < 16; off <<= 1)
                mx = fmaxf(mx, __shfl_xor(mx, off, 32));
            float nm = fmaxf(m8[vr], mx);
            float p0 = __expf(a0 - nm);
            float p1 = __expf(a1 - nm);
            float rs = p0 + p1;
#pragma unroll
            for (int off = 1; off < 16; off <<= 1)
                rs += __shfl_xor(rs, off, 32);
            cf[vr] = __expf(m8[vr] - nm);
            l8[vr] = l8[vr] * cf[vr] + rs;
            m8[vr] = nm;
            s0[vr] = p0;
            s1[vr] = p1;
        }
#pragma unroll
        for (int j = 0; j < 4; ++j)
#pragma unroll
            for (int vr = 0; vr < 8; ++vr) of[j][vr] *= cf[vr];

        // --- stage P (C layout -> A layout) through wave-private LDS ---
#pragma unroll
        for (int vr = 0; vr < 8; ++vr) {
            int row = vr + 8 * half;
            Ps[w][row][r]      = (__bf16)s0[vr];
            Ps[w][row][16 + r] = (__bf16)s1[vr];
        }
        __syncthreads();

        v16bf pa;
        {
            const __bf16* p = &Ps[w][r][8 * half];
            v8bf lo = *(const v8bf*)p;
            v8bf hi = *(const v8bf*)(p + 16);
#pragma unroll
            for (int e = 0; e < 8; ++e) { pa[e] = lo[e]; pa[8 + e] = hi[e]; }
        }
        // --- P @ V : B frags from transposed V, contraction = key ---
#pragma unroll
        for (int j = 0; j < 4; ++j) {
            v16bf vb;
            const __bf16* p = &Vt[j * 16 + r][8 * half];
            v8bf lo = *(const v8bf*)p;
            v8bf hi = *(const v8bf*)(p + 16);
#pragma unroll
            for (int e = 0; e < 8; ++e) { vb[e] = lo[e]; vb[8 + e] = hi[e]; }
            of[j] = __builtin_amdgcn_wmma_f32_16x16x32_bf16(false, pa, false, vb,
                                                            (short)0, of[j], false, false);
        }
        __syncthreads();
    }

    // --- finalize + write o [B, N, H, DH] bf16 ---
    int b = bh >> 4, h = bh & 15;
#pragma unroll
    for (int j = 0; j < 4; ++j)
#pragma unroll
        for (int vr = 0; vr < 8; ++vr) {
            int q  = qt * 128 + w * 16 + vr + 8 * half;
            int dh = j * 16 + r;
            float v = of[j][vr] / l8[vr];
            o[(((size_t)b * 1024 + q) * 16 + h) * 64 + dh] = (__bf16)v;
        }
}

// ---------------------------------------------------------------------------
// Host launcher
// ---------------------------------------------------------------------------
extern "C" void kernel_launch(void* const* d_in, const int* in_sizes, int n_in,
                              void* d_out, int out_size, void* d_ws, size_t ws_size,
                              hipStream_t stream) {
    (void)in_sizes; (void)n_in; (void)out_size; (void)ws_size;
    const float* x      = (const float*)d_in[0];
    const float* ln1_g  = (const float*)d_in[1];
    const float* ln1_b  = (const float*)d_in[2];
    const float* w_qkv  = (const float*)d_in[3];
    const float* w_out  = (const float*)d_in[4];
    const float* b_out  = (const float*)d_in[5];
    const float* phase  = (const float*)d_in[6];
    const float* ln2_g  = (const float*)d_in[7];
    const float* ln2_b  = (const float*)d_in[8];
    const float* w1     = (const float*)d_in[9];
    const float* b1     = (const float*)d_in[10];
    const float* w2     = (const float*)d_in[11];
    const float* b2     = (const float*)d_in[12];
    float* out = (float*)d_out;

    constexpr size_t BT = 4096;       // B*N tokens
    constexpr size_t Dm = 1024;
    constexpr size_t I3 = 3072;
    constexpr size_t MLPd = 4096;

    char* ws = (char*)d_ws;
    size_t off = 0;
    auto carve = [&](size_t bytes) -> void* {
        void* p = ws + off;
        off += (bytes + 255) & ~(size_t)255;
        return p;
    };
    __bf16* h_bf    = (__bf16*)carve(BT * Dm * 2);
    __bf16* wqkv_t  = (__bf16*)carve(Dm * I3 * 2);      // [3072][1024]
    __bf16* wout_t  = (__bf16*)carve(Dm * Dm * 2);      // [1024][1024]
    __bf16* w1_t    = (__bf16*)carve(Dm * MLPd * 2);    // [4096][1024]
    __bf16* w2_t    = (__bf16*)carve(MLPd * Dm * 2);    // [1024][4096]
    float*  qkv_f   = (float*) carve(BT * I3 * 4);
    __bf16* q2_bf   = (__bf16*)carve((size_t)4 * 16 * 1024 * 128 * 2);
    __bf16* k2_bf   = (__bf16*)carve((size_t)4 * 16 * 1024 * 128 * 2);
    __bf16* vt_bf   = (__bf16*)carve((size_t)4 * 16 * 64 * 1024 * 2);
    __bf16* o_bf    = (__bf16*)carve(BT * Dm * 2);
    float*  x1_f    = (float*) carve(BT * Dm * 4);
    __bf16* h2_bf   = (__bf16*)carve(BT * Dm * 2);
    __bf16* t_bf    = (__bf16*)carve(BT * MLPd * 2);

    // weight transpose + convert (W[k][n] f32 -> Wt[n][k] bf16)
    cvt_transpose_bf16_kernel<<<dim3(I3 / 32, Dm / 32), 256, 0, stream>>>(
        w_qkv, wqkv_t, (int)Dm, (int)I3);
    cvt_transpose_bf16_kernel<<<dim3(Dm / 32, Dm / 32), 256, 0, stream>>>(
        w_out, wout_t, (int)Dm, (int)Dm);
    cvt_transpose_bf16_kernel<<<dim3(MLPd / 32, Dm / 32), 256, 0, stream>>>(
        w1, w1_t, (int)Dm, (int)MLPd);
    cvt_transpose_bf16_kernel<<<dim3(Dm / 32, MLPd / 32), 256, 0, stream>>>(
        w2, w2_t, (int)MLPd, (int)Dm);

    // LN1
    ln_bf16_kernel<<<BT, 256, 0, stream>>>(x, ln1_g, ln1_b, h_bf);

    // QKV GEMM: [4096,1024] x [1024,3072] -> f32
    gemm_bf16_kernel<<<dim3(I3 / 128, BT / 128), 256, 0, stream>>>(
        h_bf, wqkv_t, qkv_f, (__bf16*)nullptr,
        (const float*)nullptr, (const float*)nullptr, (int)BT, (int)I3, (int)Dm, 0);

    // PoPE
    pope_kernel<<<(4 * 16 * 1024 * 64) / 256, 256, 0, stream>>>(qkv_f, phase,
                                                                q2_bf, k2_bf, vt_bf);

    // Attention
    attn_kernel<<<dim3(1024 / 128, 4 * 16), 256, 0, stream>>>(q2_bf, k2_bf, vt_bf, o_bf);

    // Out-projection + residual: x1 = x + o @ w_out + b_out
    gemm_bf16_kernel<<<dim3(Dm / 128, BT / 128), 256, 0, stream>>>(
        o_bf, wout_t, x1_f, (__bf16*)nullptr, b_out, x, (int)BT, (int)Dm, (int)Dm, 2);

    // LN2
    ln_bf16_kernel<<<BT, 256, 0, stream>>>(x1_f, ln2_g, ln2_b, h2_bf);

    // MLP1 + GELU: t = gelu(h2 @ w1 + b1)  (bf16 output)
    gemm_bf16_kernel<<<dim3(MLPd / 128, BT / 128), 256, 0, stream>>>(
        h2_bf, w1_t, (float*)nullptr, t_bf, b1, (const float*)nullptr,
        (int)BT, (int)MLPd, (int)Dm, 3);

    // MLP2 + residual: out = x1 + t @ w2 + b2
    gemm_bf16_kernel<<<dim3(Dm / 128, BT / 128), 256, 0, stream>>>(
        t_bf, w2_t, out, (__bf16*)nullptr, b2, x1_f, (int)BT, (int)Dm, (int)MLPd, 2);
}
